// MessagePassingBlock_13048110646129
// MI455X (gfx1250) — matrix operand order repaired and checked by reading, not compiled, not run on hardware
//
#include <hip/hip_runtime.h>

#define HIDDEN 128
#define LN_EPS 1e-5f
#define EWAVES 4
#define NWAVES 4

typedef __attribute__((ext_vector_type(16))) __bf16 v16bf;
typedef __attribute__((ext_vector_type(8)))  __bf16 v8bf;
typedef __attribute__((ext_vector_type(8)))  float  v8f;
typedef __attribute__((ext_vector_type(4)))  float  v4f;

union AB16 { v16bf v; v8bf h[2]; };

// one WMMA with literal reuse_a flag (builtin requires constant modifiers)
#define WMMA_ACC(ACC, A, B, RA) \
  (ACC) = __builtin_amdgcn_wmma_f32_16x16x32_bf16( \
      false, (A).v, false, (B).v, (short)0, (ACC), (RA), false)

// 8 consecutive WMMAs sharing the same A operand: set reuse-A hint on all but last
#define WMMA8(ACC, A, B)           \
  do {                             \
    WMMA_ACC((ACC)[0], A, (B)[0], 1); \
    WMMA_ACC((ACC)[1], A, (B)[1], 1); \
    WMMA_ACC((ACC)[2], A, (B)[2], 1); \
    WMMA_ACC((ACC)[3], A, (B)[3], 1); \
    WMMA_ACC((ACC)[4], A, (B)[4], 1); \
    WMMA_ACC((ACC)[5], A, (B)[5], 1); \
    WMMA_ACC((ACC)[6], A, (B)[6], 1); \
    WMMA_ACC((ACC)[7], A, (B)[7], 0); \
  } while (0)

__device__ __forceinline__ int iminf(int a, int b) { return a < b ? a : b; }

// load 8 contiguous f32 (two b128 loads) and pack to 8 bf16
__device__ __forceinline__ v8bf cvt8_f32_bf16(const float* __restrict__ p) {
  v4f a = *(const v4f*)p;
  v4f b = *(const v4f*)(p + 4);
  v8bf r;
  r[0] = (__bf16)a[0]; r[1] = (__bf16)a[1]; r[2] = (__bf16)a[2]; r[3] = (__bf16)a[3];
  r[4] = (__bf16)b[0]; r[5] = (__bf16)b[1]; r[6] = (__bf16)b[2]; r[7] = (__bf16)b[3];
  return r;
}

__device__ __forceinline__ v8f v8f_zero() {
  v8f z = {0.f, 0.f, 0.f, 0.f, 0.f, 0.f, 0.f, 0.f};
  return z;
}

// -------- prep: W[K,N] f32 (row-major) -> WT[N,K] bf16 --------
__global__ void transpose_to_bf16_kernel(const float* __restrict__ W,
                                         __bf16* __restrict__ WT, int K, int N) {
  int i = blockIdx.x * blockDim.x + threadIdx.x;
  if (i >= K * N) return;
  int k = i / N, n = i - k * N;
  WT[(size_t)n * K + k] = (__bf16)W[(size_t)k * N + n];
}

__global__ void zero_f32_kernel(float* __restrict__ p, int n) {
  int i = blockIdx.x * blockDim.x + threadIdx.x;
  if (i < n) p[i] = 0.f;
}

// -------- edge kernel: gather -> MLP2(384->128->128, SiLU) -> scatter-add --------
__global__ void __launch_bounds__(32 * EWAVES)
edge_message_kernel(const float* __restrict__ h_node,
                    const float* __restrict__ h_edge,
                    const int*   __restrict__ edge_index,
                    const __bf16* __restrict__ W1T, const float* __restrict__ b1,
                    const __bf16* __restrict__ W2T, const float* __restrict__ b2,
                    float* __restrict__ agg, int nEdges) {
  __shared__ __bf16 sH[EWAVES][16][HIDDEN];    // layer-1 activations (post-SiLU)
  __shared__ __bf16 sW2[HIDDEN][HIDDEN];       // layer-2 weights [N][K], 32 KB
  __shared__ int    sSrc[EWAVES][16];
  __shared__ int    sDst[EWAVES][16];

  // block-cooperative stage of the layer-2 weight matrix into LDS
  {
    v16bf* dst = (v16bf*)&sW2[0][0];
    const v16bf* src = (const v16bf*)W2T;
    for (int i = threadIdx.x; i < (HIDDEN * HIDDEN) / 16; i += blockDim.x)
      dst[i] = src[i];
  }
  __syncthreads();

  const int wave = threadIdx.x >> 5;
  const int lane = threadIdx.x & 31;
  const int tile = blockIdx.x * EWAVES + wave;
  const int e0   = tile * 16;
  if (e0 >= nEdges) return;            // wave-uniform: EXEC stays all-ones for WMMA

  if (lane < 16) {
    int e = iminf(e0 + lane, nEdges - 1);
    sSrc[wave][lane] = edge_index[e];
    sDst[wave][lane] = edge_index[nEdges + e];
  }

  const int row  = lane & 15;          // A matrix row handled by this lane
  const int half = lane >> 4;          // lane half selects K sub-range
  const int koB  = half * 16;          // A-fragment K offset (elements)
  const int kwB  = half * 16;          // B-fragment K offset (elements)
  const int col  = row;                // B/C column within a 16-wide N tile
  const int mB   = half * 8;           // C-fragment row base

  const int eRow = iminf(e0 + row, nEdges - 1);
  const float* aBase[3] = {
    h_node + (size_t)sSrc[wave][row] * HIDDEN,   // K [0,128)
    h_node + (size_t)sDst[wave][row] * HIDDEN,   // K [128,256)
    h_edge + (size_t)eRow * HIDDEN               // K [256,384)
  };

  // ---- layer 1: [16,384] x [384,128] ----
  v8f acc[8];
#pragma unroll
  for (int n = 0; n < 8; ++n) acc[n] = v8f_zero();

#pragma unroll
  for (int ks = 0; ks < 12; ++ks) {
    const int kk = ks * 32;
    const float* sp = aBase[kk >> 7] + (kk & 127);
    AB16 a;
    a.h[0] = cvt8_f32_bf16(sp + koB);        // K = kk+koB   .. +7
    a.h[1] = cvt8_f32_bf16(sp + koB + 8);    // K = kk+koB+8 .. +15
    // preload all 8 B fragments first: one long load clause, staggered waits
    AB16 b[8];
#pragma unroll
    for (int n = 0; n < 8; ++n)
      b[n].v = *(const v16bf*)(W1T + (size_t)(n * 16 + col) * (3 * HIDDEN) + kk + kwB);
    WMMA8(acc, a, b);
  }

  // bias + SiLU -> LDS (re-layout C-fragment -> A-fragment)
#pragma unroll
  for (int n = 0; n < 8; ++n) {
    const float bb = b1[n * 16 + col];
#pragma unroll
    for (int r = 0; r < 8; ++r) {
      float x = acc[n][r] + bb;
      float s = x / (1.f + __expf(-x));      // SiLU
      sH[wave][mB + r][n * 16 + col] = (__bf16)s;
    }
  }

  // ---- layer 2: [16,128] x [128,128], weights from LDS ----
  v8f acc2[8];
#pragma unroll
  for (int n = 0; n < 8; ++n) acc2[n] = v8f_zero();

#pragma unroll
  for (int ks = 0; ks < 4; ++ks) {
    const int kk = ks * 32;
    const __bf16* hp = &sH[wave][row][kk + koB];
    AB16 a;
    a.h[0] = *(const v8bf*)hp;
    a.h[1] = *(const v8bf*)(hp + 8);
    AB16 b[8];
#pragma unroll
    for (int n = 0; n < 8; ++n)
      b[n].v = *(const v16bf*)(&sW2[n * 16 + col][kk + kwB]);
    WMMA8(acc2, a, b);
  }

  // scatter-add messages into agg (L2-resident)
#pragma unroll
  for (int n = 0; n < 8; ++n) {
    const float bb = b2[n * 16 + col];
#pragma unroll
    for (int r = 0; r < 8; ++r) {
      const int m = mB + r;
      if (e0 + m < nEdges) {
        const int node = sDst[wave][m];
        atomicAdd(&agg[(size_t)node * HIDDEN + n * 16 + col], acc2[n][r] + bb);
      }
    }
  }
}

// -------- node kernel: [h_node|agg] -> MLP2(256->128->128) -> residual -> LayerNorm --------
__global__ void __launch_bounds__(32 * NWAVES)
node_update_kernel(const float* __restrict__ h_node,
                   const float* __restrict__ agg,
                   const __bf16* __restrict__ U1T, const float* __restrict__ ub1,
                   const __bf16* __restrict__ U2T, const float* __restrict__ ub2,
                   const float* __restrict__ gamma, const float* __restrict__ beta,
                   float* __restrict__ out, int nNodes) {
  __shared__ __bf16 sH[NWAVES][16][HIDDEN];
  __shared__ float  sOut[NWAVES][16][HIDDEN];

  const int wave = threadIdx.x >> 5;
  const int lane = threadIdx.x & 31;
  const int tile = blockIdx.x * NWAVES + wave;
  const int n0   = tile * 16;
  if (n0 >= nNodes) return;

  const int row  = lane & 15;
  const int half = lane >> 4;
  const int koB  = half * 16;
  const int kwB  = half * 16;
  const int col  = row;
  const int mB   = half * 8;

  const int nodeRow = iminf(n0 + row, nNodes - 1);
  const float* aBase[2] = {
    h_node + (size_t)nodeRow * HIDDEN,   // K [0,128)
    agg    + (size_t)nodeRow * HIDDEN    // K [128,256)
  };

  // ---- layer 1: [16,256] x [256,128] ----
  v8f acc[8];
#pragma unroll
  for (int n = 0; n < 8; ++n) acc[n] = v8f_zero();

#pragma unroll
  for (int ks = 0; ks < 8; ++ks) {
    const int kk = ks * 32;
    const float* sp = aBase[kk >> 7] + (kk & 127);
    AB16 a;
    a.h[0] = cvt8_f32_bf16(sp + koB);
    a.h[1] = cvt8_f32_bf16(sp + koB + 8);
    AB16 b[8];
#pragma unroll
    for (int n = 0; n < 8; ++n)
      b[n].v = *(const v16bf*)(U1T + (size_t)(n * 16 + col) * (2 * HIDDEN) + kk + kwB);
    WMMA8(acc, a, b);
  }

#pragma unroll
  for (int n = 0; n < 8; ++n) {
    const float bb = ub1[n * 16 + col];
#pragma unroll
    for (int r = 0; r < 8; ++r) {
      float x = acc[n][r] + bb;
      float s = x / (1.f + __expf(-x));
      sH[wave][mB + r][n * 16 + col] = (__bf16)s;
    }
  }

  // ---- layer 2: [16,128] x [128,128] ----
  v8f acc2[8];
#pragma unroll
  for (int n = 0; n < 8; ++n) acc2[n] = v8f_zero();

#pragma unroll
  for (int ks = 0; ks < 4; ++ks) {
    const int kk = ks * 32;
    const __bf16* hp = &sH[wave][row][kk + koB];
    AB16 a;
    a.h[0] = *(const v8bf*)hp;
    a.h[1] = *(const v8bf*)(hp + 8);
    AB16 b[8];
#pragma unroll
    for (int n = 0; n < 8; ++n)
      b[n].v = *(const v16bf*)(U2T + (size_t)(n * 16 + col) * HIDDEN + kk + kwB);
    WMMA8(acc2, a, b);
  }

  // bias + residual -> LDS (f32)
#pragma unroll
  for (int n = 0; n < 8; ++n) {
    const float bb = ub2[n * 16 + col];
#pragma unroll
    for (int r = 0; r < 8; ++r) {
      const int m  = mB + r;
      const int nr = iminf(n0 + m, nNodes - 1);
      float v = acc2[n][r] + bb + h_node[(size_t)nr * HIDDEN + n * 16 + col];
      sOut[wave][m][n * 16 + col] = v;
    }
  }

  // ---- LayerNorm: 32 lanes x 4 cols per row, wave32 shuffles ----
  const v4f g4 = *(const v4f*)&gamma[lane * 4];
  const v4f be4 = *(const v4f*)&beta[lane * 4];
  for (int m = 0; m < 16; ++m) {
    if (n0 + m >= nNodes) break;
    v4f x = *(const v4f*)&sOut[wave][m][lane * 4];
    float s = x[0] + x[1] + x[2] + x[3];
#pragma unroll
    for (int off = 16; off > 0; off >>= 1) s += __shfl_xor(s, off, 32);
    const float mean = s * (1.f / HIDDEN);
    v4f d = x - mean;
    float q = d[0]*d[0] + d[1]*d[1] + d[2]*d[2] + d[3]*d[3];
#pragma unroll
    for (int off = 16; off > 0; off >>= 1) q += __shfl_xor(q, off, 32);
    const float rstd = rsqrtf(q * (1.f / HIDDEN) + LN_EPS);
    v4f y = d * rstd * g4 + be4;
    *(v4f*)&out[(size_t)(n0 + m) * HIDDEN + lane * 4] = y;
  }
}

extern "C" void kernel_launch(void* const* d_in, const int* in_sizes, int n_in,
                              void* d_out, int out_size, void* d_ws, size_t ws_size,
                              hipStream_t stream) {
  const float* h_node   = (const float*)d_in[0];
  const float* h_edge   = (const float*)d_in[1];
  const int*   edge_idx = (const int*)d_in[2];
  const float* msg_W1   = (const float*)d_in[3];
  const float* msg_b1   = (const float*)d_in[4];
  const float* msg_W2   = (const float*)d_in[5];
  const float* msg_b2   = (const float*)d_in[6];
  const float* upd_W1   = (const float*)d_in[7];
  const float* upd_b1   = (const float*)d_in[8];
  const float* upd_W2   = (const float*)d_in[9];
  const float* upd_b2   = (const float*)d_in[10];
  const float* ln_gamma = (const float*)d_in[11];
  const float* ln_beta  = (const float*)d_in[12];

  const int N = in_sizes[0] / HIDDEN;   // 50000
  const int E = in_sizes[2] / 2;        // 600000

  // workspace layout: bf16 transposed weights, then f32 agg buffer
  char* ws = (char*)d_ws;
  __bf16* W1T = (__bf16*)(ws + 0);        // 384*128 bf16 = 98304 B
  __bf16* W2T = (__bf16*)(ws + 98304);    // 128*128 bf16 = 32768 B
  __bf16* U1T = (__bf16*)(ws + 131072);   // 256*128 bf16 = 65536 B
  __bf16* U2T = (__bf16*)(ws + 196608);   // 128*128 bf16 = 32768 B
  float*  agg = (float*)(ws + 262144);    // N*128 f32

  // 1) weight prep (convert + transpose to [N,K] bf16)
  transpose_to_bf16_kernel<<<(3 * HIDDEN * HIDDEN + 255) / 256, 256, 0, stream>>>(
      msg_W1, W1T, 3 * HIDDEN, HIDDEN);
  transpose_to_bf16_kernel<<<(HIDDEN * HIDDEN + 255) / 256, 256, 0, stream>>>(
      msg_W2, W2T, HIDDEN, HIDDEN);
  transpose_to_bf16_kernel<<<(2 * HIDDEN * HIDDEN + 255) / 256, 256, 0, stream>>>(
      upd_W1, U1T, 2 * HIDDEN, HIDDEN);
  transpose_to_bf16_kernel<<<(HIDDEN * HIDDEN + 255) / 256, 256, 0, stream>>>(
      upd_W2, U2T, HIDDEN, HIDDEN);

  // 2) zero the aggregation buffer (workspace is poisoned)
  zero_f32_kernel<<<(N * HIDDEN + 255) / 256, 256, 0, stream>>>(agg, N * HIDDEN);

  // 3) fused edge message + scatter-add
  const int eTiles = (E + 15) / 16;
  edge_message_kernel<<<(eTiles + EWAVES - 1) / EWAVES, 32 * EWAVES, 0, stream>>>(
      h_node, h_edge, edge_idx, W1T, msg_b1, W2T, msg_b2, agg, E);

  // 4) fused node update + residual + layernorm
  const int nTiles = (N + 15) / 16;
  node_update_kernel<<<(nTiles + NWAVES - 1) / NWAVES, 32 * NWAVES, 0, stream>>>(
      h_node, agg, U1T, upd_b1, U2T, upd_b2, ln_gamma, ln_beta, (float*)d_out, N);
}
